// CausalSelfAttention_68874095559207
// MI455X (gfx1250) — matrix operand order repaired and checked by reading, not compiled, not run on hardware
//
#include <hip/hip_runtime.h>

typedef __attribute__((ext_vector_type(16))) _Float16 v16h;
typedef __attribute__((ext_vector_type(8)))  float    v8f;
typedef __attribute__((ext_vector_type(8)))  _Float16 h8;   // 16-byte chunk
typedef __attribute__((ext_vector_type(4)))  int      v4i;  // 16-byte chunk

#define D_MODEL 1024
#define N_HEAD  16
#define DH      64
#define BATCH   4
#define SEQ     2048
#define ROWS    (BATCH * SEQ)   // 8192

// ---------- CDNA5 async global->LDS staging (GLOBAL_LOAD_ASYNC_TO_LDS_B128) ----

#if defined(__has_builtin)
#  if __has_builtin(__builtin_amdgcn_global_load_async_to_lds_b128)
#    define USE_ASYNC_LDS 1
#  endif
#endif
#ifndef USE_ASYNC_LDS
#  define USE_ASYNC_LDS 0
#endif

__device__ __forceinline__ void cp16(_Float16* l, const _Float16* g) {
#if USE_ASYNC_LDS
  __builtin_amdgcn_global_load_async_to_lds_b128(
      (__attribute__((address_space(1))) v4i*)g,
      (__attribute__((address_space(3))) v4i*)l, 0, 0);
#else
  *(h8*)l = *(const h8*)g;
#endif
}

// Wait until at most `n` async ops remain outstanding (loads complete in order,
// so n==6 means "the 6 most recently issued may still be in flight").
#if USE_ASYNC_LDS
#  if __has_builtin(__builtin_amdgcn_s_wait_asynccnt)
#    define WAIT_ASYNC(n) __builtin_amdgcn_s_wait_asynccnt(n)
#  else
#    define WAIT_ASYNC(n) asm volatile("s_wait_asynccnt " #n ::: "memory")
#  endif
#else
#  define WAIT_ASYNC(n) ((void)0)
#endif

// ---------- WMMA fragment loaders (contiguous k = 16*half + s, same in A and B) -----------

// A: 16x32 (MxK) f16, row-major source. Per lane: 16 halfs, contiguous per half-K.
__device__ __forceinline__ v16h ldA(const _Float16* src, int ld) {
  const int lane = threadIdx.x & 31;
  const int half = lane >> 4, m = lane & 15;
  v16h a;
#pragma unroll
  for (int v = 0; v < 8; ++v) {
    const int kb = half * 16 + 2 * v;
    a[2 * v]     = src[m * ld + kb];
    a[2 * v + 1] = src[m * ld + kb + 1];
  }
  return a;
}

// B: 32x16 (KxN) where SOURCE is N-major: logical B[k][n] = src[n*ld + k].
// Per lane the 16 halfs are contiguous -> merges into b128 loads.
__device__ __forceinline__ v16h ldBt(const _Float16* src, int ld) {
  const int lane = threadIdx.x & 31;
  const int half = lane >> 4, n = lane & 15;
  v16h b;
#pragma unroll
  for (int v = 0; v < 8; ++v) {
    const int k = half * 16 + 2 * v;
    b[2 * v]     = src[n * ld + k];
    b[2 * v + 1] = src[n * ld + k + 1];
  }
  return b;
}

__device__ __forceinline__ v8f wmma_f16(v16h a, v16h b, v8f c) {
  return __builtin_amdgcn_wmma_f32_16x16x32_f16(false, a, false, b, (short)0, c,
                                                false, false);
}

// ---------- conversion prepass -------------------------------------------------

__global__ void cvt_f32_f16(const float* __restrict__ src,
                            _Float16* __restrict__ dst, int n) {
  int i = blockIdx.x * blockDim.x + threadIdx.x;
  const int stride = gridDim.x * blockDim.x;
  for (; i < n; i += stride) dst[i] = (_Float16)src[i];
}

// src: [K][N] row-major f32 -> dst: [N][K] f16 (N-major weights for the GEMMs)
__global__ void cvt_f32_f16_T(const float* __restrict__ src,
                              _Float16* __restrict__ dst, int K, int N) {
  int i = blockIdx.x * blockDim.x + threadIdx.x;
  const int stride = gridDim.x * blockDim.x;
  const int total = K * N;
  for (; i < total; i += stride) {
    const int k = i / N, n = i - k * N;
    dst[(size_t)n * K + k] = (_Float16)src[i];
  }
}

// ---------- shared GEMM core ---------------------------------------------------
// Block tile 64x128, BK=32, 4 waves (2x2), wave 32x64.
// Double-buffered LDS; async prefetch depth 1 overlaps with WMMA.

#define LDP  40            // padded LDS row (halfs); 80B stride keeps 16B align
#define ASZ  (64 * LDP)    // one A buffer
#define BSZ  (128 * LDP)   // one B buffer
#define NT   (D_MODEL / 32)

struct GemmAcc { v8f a[2][4]; };

__device__ __forceinline__ void issue_tile(const _Float16* __restrict__ Arm,
                                           const _Float16* __restrict__ BT,
                                           int rowBase, int colBase, int k0,
                                           _Float16* As, _Float16* Bs) {
  const int tid = threadIdx.x;
#pragma unroll
  for (int ii = 0; ii < 2; ++ii) {           // A tile: 64 rows x 32 K
    const int i = tid + ii * 128;
    const int r = i >> 2, c = (i & 3) * 8;
    cp16(&As[r * LDP + c], &Arm[(size_t)(rowBase + r) * D_MODEL + k0 + c]);
  }
#pragma unroll
  for (int ii = 0; ii < 4; ++ii) {           // B tile: 128 cols x 32 K, N-major
    const int i = tid + ii * 128;
    const int nn = i >> 2, c = (i & 3) * 8;
    cp16(&Bs[nn * LDP + c], &BT[(size_t)(colBase + nn) * D_MODEL + k0 + c]);
  }
}

__device__ __forceinline__ void gemm_core(const _Float16* __restrict__ Arm,
                                          const _Float16* __restrict__ BT,
                                          int rowBase, int colBase,
                                          _Float16* As, _Float16* Bs,
                                          GemmAcc& acc) {
  const int tid = threadIdx.x;
  const int wave = tid >> 5;
  const int wm = wave >> 1, wn = wave & 1;

  issue_tile(Arm, BT, rowBase, colBase, 0, As, Bs);   // prologue: tile 0 -> buf 0

  for (int it = 0; it < NT; ++it) {
    const int cur = it & 1;
    _Float16* Ac = As + cur * ASZ;
    _Float16* Bc = Bs + cur * BSZ;
    if (it + 1 < NT) {
      // Prefetch next tile into the other buffer (last read in the previous iteration,
      // reads completed at the previous end-of-iteration barrier).
      issue_tile(Arm, BT, rowBase, colBase, (it + 1) * 32,
                 As + (1 - cur) * ASZ, Bs + (1 - cur) * BSZ);
      WAIT_ASYNC(6);   // 6 newest (next tile) may remain in flight; tile `it` done
    } else {
      WAIT_ASYNC(0);
    }
    __syncthreads();   // all waves' tile-`it` data visible in LDS

    v16h a0 = ldA(Ac + (wm * 32) * LDP, LDP);
    v16h a1 = ldA(Ac + (wm * 32 + 16) * LDP, LDP);
#pragma unroll
    for (int j = 0; j < 4; ++j) {
      v16h b = ldBt(Bc + (wn * 64 + 16 * j) * LDP, LDP);
      acc.a[0][j] = wmma_f16(a0, b, acc.a[0][j]);
      acc.a[1][j] = wmma_f16(a1, b, acc.a[1][j]);
    }
    __syncthreads();   // reads of buf `cur` done before it is overwritten
  }
}

// ---------- Kernel 1: QKV projection + scatter ---------------------------------
// qkv[8192,3072] = Xh * WqkvT^T + bias. sec = colBase>>10 is BLOCK-UNIFORM.
// Q,K -> [B,H,T,DH] f16 (Q scaled by 1/sqrt(DH)); V -> [B,H,DH,T] f16 (transposed).

__global__ void __launch_bounds__(128)
qkv_gemm(const _Float16* __restrict__ Xh, const _Float16* __restrict__ WT,
         const float* __restrict__ bias,
         _Float16* __restrict__ Qh, _Float16* __restrict__ Kh,
         _Float16* __restrict__ Vt) {
  __shared__ _Float16 As[2 * ASZ];
  __shared__ _Float16 Bs[2 * BSZ];
  const int rowBase = blockIdx.y * 64;
  const int colBase = blockIdx.x * 128;

  GemmAcc acc = {};
  gemm_core(Xh, WT, rowBase, colBase, As, Bs, acc);

  const int tid = threadIdx.x;
  const int wave = tid >> 5, wm = wave >> 1, wn = wave & 1;
  const int lane = tid & 31, half = lane >> 4, n = lane & 15;
  const int sec = colBase >> 10;               // 0=q 1=k 2=v, uniform per block

  if (sec == 2) {                              // V: transposed store [B,H,DH,T]
#pragma unroll
    for (int i = 0; i < 2; ++i)
#pragma unroll
      for (int j = 0; j < 4; ++j)
#pragma unroll
        for (int r = 0; r < 8; ++r) {
          const int row = rowBase + wm * 32 + i * 16 + r + 8 * half;
          const int col = colBase + wn * 64 + j * 16 + n;
          const float val = acc.a[i][j][r] + bias[col];
          const int c = col & 1023, h = c >> 6, d = c & 63;
          const int b = row >> 11, t = row & 2047;
          Vt[(((size_t)(b * N_HEAD + h) * DH) + d) * SEQ + t] = (_Float16)val;
        }
  } else {                                     // Q or K: [B,H,T,DH]
    _Float16* dst = sec ? Kh : Qh;
    const float scl = sec ? 1.0f : 0.125f;     // fold 1/sqrt(DH) into Q
#pragma unroll
    for (int i = 0; i < 2; ++i)
#pragma unroll
      for (int j = 0; j < 4; ++j)
#pragma unroll
        for (int r = 0; r < 8; ++r) {
          const int row = rowBase + wm * 32 + i * 16 + r + 8 * half;
          const int col = colBase + wn * 64 + j * 16 + n;
          const float val = (acc.a[i][j][r] + bias[col]) * scl;
          const int c = col & 1023, h = c >> 6, d = c & 63;
          const int b = row >> 11, t = row & 2047;
          dst[(((size_t)(b * N_HEAD + h) * SEQ) + t) * DH + d] = (_Float16)val;
        }
  }
}

// ---------- Kernel 2: causal flash attention -----------------------------------
// 4 independent waves / block; wave owns 16 query rows; 32 keys per iteration.
// K is [T,DH] (B-frags via ldBt, contiguous); V is [DH,T] (also ldBt, contiguous).

__global__ void __launch_bounds__(128)
flash_attn(const _Float16* __restrict__ Qh, const _Float16* __restrict__ Kh,
           const _Float16* __restrict__ Vt, _Float16* __restrict__ Yh) {
  __shared__ _Float16 Ps[4 * 16 * 32];
  const int tid = threadIdx.x;
  const int wave = tid >> 5;
  const int lane = tid & 31, half = lane >> 4, n = lane & 15;
  const int bh = blockIdx.y;
  const int b = bh >> 4, h = bh & 15;
  const int qw = blockIdx.x * 64 + wave * 16;
  const _Float16* Qb = Qh + (size_t)bh * SEQ * DH;
  const _Float16* Kb = Kh + (size_t)bh * SEQ * DH;
  const _Float16* Vb = Vt + (size_t)bh * DH * SEQ;
  _Float16* P = Ps + wave * 16 * 32;

  const v16h aQ0 = ldA(Qb + (size_t)qw * DH, DH);       // dh 0..31
  const v16h aQ1 = ldA(Qb + (size_t)qw * DH + 32, DH);  // dh 32..63

  v8f o[4] = {};
  float mrun[8], lrun[8];
#pragma unroll
  for (int r = 0; r < 8; ++r) { mrun[r] = -3.0e38f; lrun[r] = 0.0f; }

  for (int key0 = 0; key0 < qw + 16; key0 += 32) {
    v8f s0 = {}, s1 = {};
    v16h bk;
    bk = ldBt(Kb + (size_t)key0 * DH, DH);              s0 = wmma_f16(aQ0, bk, s0);
    bk = ldBt(Kb + (size_t)key0 * DH + 32, DH);         s0 = wmma_f16(aQ1, bk, s0);
    bk = ldBt(Kb + (size_t)(key0 + 16) * DH, DH);       s1 = wmma_f16(aQ0, bk, s1);
    bk = ldBt(Kb + (size_t)(key0 + 16) * DH + 32, DH);  s1 = wmma_f16(aQ1, bk, s1);

#pragma unroll
    for (int r = 0; r < 8; ++r) {
      const int q = qw + r + 8 * half;
      float e0 = s0[r]; if (key0 + n > q)      e0 = -3.0e38f;
      float e1 = s1[r]; if (key0 + 16 + n > q) e1 = -3.0e38f;
      float mx = fmaxf(e0, e1);
      mx = fmaxf(mx, __shfl_xor(mx, 1, 32));
      mx = fmaxf(mx, __shfl_xor(mx, 2, 32));
      mx = fmaxf(mx, __shfl_xor(mx, 4, 32));
      mx = fmaxf(mx, __shfl_xor(mx, 8, 32));
      const float mnew  = fmaxf(mrun[r], mx);
      const float alpha = __expf(mrun[r] - mnew);
      const float p0 = __expf(e0 - mnew);
      const float p1 = __expf(e1 - mnew);
      float rs = p0 + p1;
      rs += __shfl_xor(rs, 1, 32);
      rs += __shfl_xor(rs, 2, 32);
      rs += __shfl_xor(rs, 4, 32);
      rs += __shfl_xor(rs, 8, 32);
      lrun[r] = lrun[r] * alpha + rs;
      mrun[r] = mnew;
#pragma unroll
      for (int j = 0; j < 4; ++j) o[j][r] *= alpha;
      P[(r + 8 * half) * 32 + n]      = (_Float16)p0;
      P[(r + 8 * half) * 32 + 16 + n] = (_Float16)p1;
    }
    asm volatile("s_wait_dscnt 0" ::: "memory");  // per-wave LDS RAW
    const v16h aP = ldA(P, 32);
#pragma unroll
    for (int j = 0; j < 4; ++j) {
      v16h bv = ldBt(Vb + (size_t)(j * 16) * SEQ + key0, SEQ);
      o[j] = wmma_f16(aP, bv, o[j]);
    }
  }

#pragma unroll
  for (int r = 0; r < 8; ++r) {
    const int q = qw + r + 8 * half;
    const float inv = 1.0f / lrun[r];
#pragma unroll
    for (int j = 0; j < 4; ++j) {
      const float val = o[j][r] * inv;
      Yh[((size_t)(b * SEQ + q)) * D_MODEL + h * DH + j * 16 + n] = (_Float16)val;
    }
  }
}

// ---------- Kernel 3: output projection ----------------------------------------

__global__ void __launch_bounds__(128)
out_gemm(const _Float16* __restrict__ Yh, const _Float16* __restrict__ WT,
         const float* __restrict__ bias, float* __restrict__ out) {
  __shared__ _Float16 As[2 * ASZ];
  __shared__ _Float16 Bs[2 * BSZ];
  const int rowBase = blockIdx.y * 64;
  const int colBase = blockIdx.x * 128;

  GemmAcc acc = {};
  gemm_core(Yh, WT, rowBase, colBase, As, Bs, acc);

  const int tid = threadIdx.x;
  const int wave = tid >> 5, wm = wave >> 1, wn = wave & 1;
  const int lane = tid & 31, half = lane >> 4, n = lane & 15;
#pragma unroll
  for (int i = 0; i < 2; ++i)
#pragma unroll
    for (int j = 0; j < 4; ++j)
#pragma unroll
      for (int r = 0; r < 8; ++r) {
        const int row = rowBase + wm * 32 + i * 16 + r + 8 * half;
        const int col = colBase + wn * 64 + j * 16 + n;
        out[(size_t)row * D_MODEL + col] = acc.a[i][j][r] + bias[col];
      }
}

// ---------- Launcher -----------------------------------------------------------

extern "C" void kernel_launch(void* const* d_in, const int* in_sizes, int n_in,
                              void* d_out, int out_size, void* d_ws, size_t ws_size,
                              hipStream_t stream) {
  const float* x    = (const float*)d_in[0];
  const float* Wqkv = (const float*)d_in[1];
  const float* bqkv = (const float*)d_in[2];
  const float* Wout = (const float*)d_in[3];
  const float* bout = (const float*)d_in[4];
  float* out = (float*)d_out;

  char* ws = (char*)d_ws;
  const size_t MB = 1024u * 1024u;
  _Float16* xh     = (_Float16*)(ws + 0 * MB);   // 16 MiB  [8192,1024]
  _Float16* wqkvT  = (_Float16*)(ws + 16 * MB);  // 6 MiB   [3072,1024] N-major
  _Float16* woutT  = (_Float16*)(ws + 22 * MB);  // 2 MiB   [1024,1024] N-major
  _Float16* Qh     = (_Float16*)(ws + 24 * MB);  // 16 MiB  [B,H,T,DH]
  _Float16* Kh     = (_Float16*)(ws + 40 * MB);  // 16 MiB  [B,H,T,DH]
  _Float16* Vt     = (_Float16*)(ws + 56 * MB);  // 16 MiB  [B,H,DH,T]
  _Float16* Yh     = (_Float16*)(ws + 72 * MB);  // 16 MiB  [B,T,C]

  cvt_f32_f16<<<4096, 256, 0, stream>>>(x, xh, ROWS * D_MODEL);
  cvt_f32_f16_T<<<4096, 256, 0, stream>>>(Wqkv, wqkvT, D_MODEL, 3 * D_MODEL);
  cvt_f32_f16_T<<<2048, 256, 0, stream>>>(Wout, woutT, D_MODEL, D_MODEL);

  qkv_gemm<<<dim3(24, 128), 128, 0, stream>>>(xh, wqkvT, bqkv, Qh, Kh, Vt);
  flash_attn<<<dim3(32, 64), 128, 0, stream>>>(Qh, Kh, Vt, Yh);
  out_gemm<<<dim3(8, 128), 128, 0, stream>>>(Yh, woutT, bout, out);
}
